// DeformConvF2F_55104430408273
// MI455X (gfx1250) — matrix operand order repaired and checked by reading, not compiled, not run on hardware
//
#include <hip/hip_runtime.h>
#include <hip/hip_fp16.h>

typedef __attribute__((ext_vector_type(16))) _Float16 v16h;
typedef __attribute__((ext_vector_type(8)))  float    v8f;

#define IMG_H  64
#define IMG_W  64
#define IMG_HW 4096
#define NBATCH 4

// ---------------------------------------------------------------------------
// Weight conversion: fp32 [cout][cin][K][K]  ->  f16 [tap][cout][cin]
// ---------------------------------------------------------------------------
__global__ void cvt_weights_kernel(const float* __restrict__ w,
                                   _Float16* __restrict__ wf,
                                   int cout, int cin, int kk) {
  int e = blockIdx.x * blockDim.x + threadIdx.x;
  int total = cout * cin * kk;
  if (e >= total) return;
  int t = e % kk;
  int c = (e / kk) % cin;
  int o = e / (kk * cin);
  wf[((size_t)t * cout + o) * cin + c] = (_Float16)w[e];
}

// ---------------------------------------------------------------------------
// One deformable-conv layer. Workgroup = 256 threads (8 wave32) = 32 pixels
// (half of one image row). Wave w owns MT cout-tiles x NT(=2) pixel-tiles.
// Software-pipelined: double-buffered LDS staging, one barrier per 32-wide
// cin chunk; v_wmma_f32_16x16x32_f16 with f32 accumulation.
// ---------------------------------------------------------------------------
template <int TAPS, int MT, int NT>
__global__ __launch_bounds__(256)
void deform_layer_kernel(const float* __restrict__ x,     // [N][cin][H][W] fp32
                         const float* __restrict__ off,   // [N][2*TAPS][H][W]
                         const _Float16* __restrict__ wt, // [TAPS][cout][cin] f16
                         const float* __restrict__ bias,  // [cout]
                         float* __restrict__ out,         // [N][cout][H][W] fp32
                         int cin, int cout, int K, int pad) {
  __shared__ __align__(16) int      s_idx[TAPS][32][4]; // corner idx (h*W+w), 0 if OOB
  __shared__ __align__(16) float    s_w[TAPS][32][4];   // corner weights, 0 if OOB
  __shared__ _Float16 s_B[2][32][34];                   // [buf][pixel][cin], 34-half
                                                        // rows = 17 banks, no conflicts

  const int tid    = threadIdx.x;
  const int lane   = tid & 31;
  const int wave   = tid >> 5;
  const int hiHalf = (lane >= 16) ? 1 : 0;
  const int sp     = tid & 31;   // staging pixel owned by this thread (fixed)
  const int sk0    = tid >> 5;   // staging channel base 0..7 (fixed)

  const int pixBase  = blockIdx.x * 32;          // flat pixel over N*H*W
  const int n        = pixBase >> 12;            // / IMG_HW
  const int pixInImg = pixBase & (IMG_HW - 1);
  const int h  = pixInImg >> 6;
  const int wp = pixInImg & (IMG_W - 1);

  // ---- per-(tap,pixel) bilinear metadata, shared across all channels -------
  for (int idx = tid; idx < TAPS * 32; idx += 256) {
    int t = idx >> 5;
    int p = idx & 31;
    int kh = t / K, kw = t % K;
    size_t obase = (((size_t)n * (2 * TAPS) + 2 * t) * IMG_H + h) * IMG_W + (wp + p);
    float dy = off[obase];
    float dx = off[obase + (size_t)IMG_H * IMG_W];
    float py = (float)(h + kh - pad) + dy;
    float px = (float)(wp + p + kw - pad) + dx;
    float fy0 = floorf(py), fx0 = floorf(px);
    int y0 = (int)fy0, x0 = (int)fx0;
    float ly = py - fy0, lx = px - fx0;
    float wy[2] = {1.f - ly, ly};
    float wx[2] = {1.f - lx, lx};
#pragma unroll
    for (int c = 0; c < 4; ++c) {
      int yc = y0 + (c >> 1);
      int xc = x0 + (c & 1);
      bool valid = (yc >= 0) && (yc < IMG_H) && (xc >= 0) && (xc < IMG_W);
      s_idx[t][p][c] = valid ? (yc * IMG_W + xc) : 0;
      s_w[t][p][c]   = valid ? wy[c >> 1] * wx[c & 1] : 0.f;
    }
  }

  // channel base pointer for this thread's staging work (channel = sk0 + 8j + 32*kc)
  const float* const xbase = x + ((size_t)n * cin + sk0) * IMG_HW;

  // stage 32 channels x 32 pixels of bilinear samples for (tap t, chunk kc)
  auto stageChunk = [&](int t, int kc, int buf) {
    const int4   ip = *(const int4*)(&s_idx[t][sp][0]);
    const float4 wc = *(const float4*)(&s_w[t][sp][0]);
    const float* xk = xbase + (size_t)(kc << 5) * IMG_HW;
#pragma unroll
    for (int j = 0; j < 4; ++j) {
      const float* ximg = xk + (size_t)(8 * j) * IMG_HW;
      float s = wc.x * ximg[ip.x] + wc.y * ximg[ip.y]
              + wc.z * ximg[ip.z] + wc.w * ximg[ip.w];
      s_B[buf][sp][sk0 + 8 * j] = (_Float16)s;
    }
  };

  v8f acc[MT][NT];
  const v8f vzero = {0.f, 0.f, 0.f, 0.f, 0.f, 0.f, 0.f, 0.f};
#pragma unroll
  for (int i = 0; i < MT; ++i)
#pragma unroll
    for (int j = 0; j < NT; ++j) acc[i][j] = vzero;

  const int nkc   = cin >> 5;
  const int total = TAPS * nkc;

  __syncthreads();           // metadata visible
  stageChunk(0, 0, 0);       // prologue
  __syncthreads();

  int t = 0, kc = 0;
  for (int chunk = 0; chunk < total; ++chunk) {
    const int cur = chunk & 1;

    // ---- B fragments (32x16 each) from LDS buffer `cur`
    v16h bfrag[NT];
#pragma unroll
    for (int nt = 0; nt < NT; ++nt) {
      const uint32_t* brow =
          (const uint32_t*)&s_B[cur][(lane & 15) + 16 * nt][hiHalf * 16];
      uint32_t* bf = (uint32_t*)&bfrag[nt];
#pragma unroll
      for (int r = 0; r < 8; ++r) bf[r] = brow[r];
    }

    // ---- A fragments from global f16 weights; MT x NT WMMAs
#pragma unroll
    for (int i = 0; i < MT; ++i) {
      int m = (wave + 8 * i) * 16 + (lane & 15);
      const uint32_t* arow =
          (const uint32_t*)(wt + ((size_t)t * cout + m) * cin + (kc << 5) + hiHalf * 16);
      v16h afrag;
      uint32_t* af = (uint32_t*)&afrag;
#pragma unroll
      for (int r = 0; r < 8; ++r)
        af[r] = arow[r];
#pragma unroll
      for (int nt = 0; nt < NT; ++nt)
        acc[i][nt] = __builtin_amdgcn_wmma_f32_16x16x32_f16(
            false, afrag, false, bfrag[nt], (short)0, acc[i][nt], false, false);
    }

    // ---- stage next chunk into the other buffer (overlaps with WMMA issue)
    int t2 = t, kc2 = kc + 1;
    if (kc2 == nkc) { kc2 = 0; t2 = t + 1; }
    if (chunk + 1 < total) stageChunk(t2, kc2, cur ^ 1);
    t = t2; kc = kc2;
    __syncthreads();
  }

  // ---- epilogue: bias + ReLU, fp32 NCHW (C layout: VGPR r -> M = r + 8*hi)
#pragma unroll
  for (int i = 0; i < MT; ++i) {
    int mBase = (wave + 8 * i) * 16 + hiHalf * 8;
#pragma unroll
    for (int nt = 0; nt < NT; ++nt) {
      int p = 16 * nt + (lane & 15);
      float* optr = out + (size_t)n * cout * IMG_HW + pixInImg + p;
#pragma unroll
      for (int r = 0; r < 8; ++r) {
        int m = mBase + r;
        float v = acc[i][nt][r] + bias[m];
        optr[(size_t)m * IMG_HW] = v > 0.f ? v : 0.f;
      }
    }
  }
}

// ---------------------------------------------------------------------------
extern "C" void kernel_launch(void* const* d_in, const int* in_sizes, int n_in,
                              void* d_out, int out_size, void* d_ws, size_t ws_size,
                              hipStream_t stream) {
  (void)in_sizes; (void)n_in; (void)out_size; (void)ws_size;

  static const int dims[8][4] = {  // {cin, cout, K, pad}
      {512, 256, 1, 0}, {256, 128, 3, 1}, {128, 128, 3, 1}, {128, 128, 3, 1},
      {128, 128, 3, 1}, {128, 128, 3, 1}, {128, 128, 3, 1}, {128, 128, 3, 1}};

  const float* x = (const float*)d_in[0];
  const float* off[8]; const float* w[8]; const float* b[8];
  for (int i = 0; i < 8; ++i) {
    off[i] = (const float*)d_in[1 + i];
    w[i]   = (const float*)d_in[9 + 2 * i];
    b[i]   = (const float*)d_in[10 + 2 * i];
  }

  // workspace: two fp32 ping-pong activation buffers (max 256 ch) + f16 weights
  char* ws = (char*)d_ws;
  const size_t actBytes = (size_t)NBATCH * 256 * IMG_HW * sizeof(float);  // 16 MB
  float* actA = (float*)ws;
  float* actB = (float*)(ws + actBytes);
  _Float16* wf = (_Float16*)(ws + 2 * actBytes);

  size_t woff[8]; size_t accum = 0;
  for (int i = 0; i < 8; ++i) {
    woff[i] = accum;
    accum += (size_t)dims[i][1] * dims[i][0] * dims[i][2] * dims[i][2];
  }

  for (int i = 0; i < 8; ++i) {
    int cin = dims[i][0], cout = dims[i][1], K = dims[i][2];
    int total = cout * cin * K * K;
    cvt_weights_kernel<<<(total + 255) / 256, 256, 0, stream>>>(
        w[i], wf + woff[i], cout, cin, K * K);
  }

  const int grid = NBATCH * IMG_HW / 32;  // 512 workgroups, 8 waves each
  const float* cur = x;
  for (int l = 0; l < 8; ++l) {
    int cin = dims[l][0], cout = dims[l][1], K = dims[l][2], pad = dims[l][3];
    float* dst = (l == 7) ? (float*)d_out : (((l & 1) == 0) ? actA : actB);
    if (K == 1) {
      deform_layer_kernel<1, 2, 2><<<grid, 256, 0, stream>>>(
          cur, off[l], wf + woff[l], b[l], dst, cin, cout, K, pad);
    } else {
      deform_layer_kernel<9, 1, 2><<<grid, 256, 0, stream>>>(
          cur, off[l], wf + woff[l], b[l], dst, cin, cout, K, pad);
    }
    cur = dst;
  }
}
